// AffineCoupling_1451698946354
// MI455X (gfx1250) — compile-verified, matrix operand order checked by reading, not run on hardware
//
#include <hip/hip_runtime.h>

// ---------------------------------------------------------------------------
// Affine coupling (RealNVP) forward for MI455X (gfx1250, wave32, WMMA).
// Compute-bound (~688 GFLOP vs ~0.4 GB HBM) -> bf16 WMMA with f32 accumulate.
// Global->LDS tile movement uses gfx1250 async-LDS loads (ASYNCcnt) to avoid
// VGPR staging (earlier revision spilled staging regs to scratch).
// ---------------------------------------------------------------------------

typedef __attribute__((ext_vector_type(16))) __bf16 v16bf;
typedef __attribute__((ext_vector_type(8)))  float  v8f;
typedef int v4i_gcc __attribute__((vector_size(16)));   // matches builtin 'V4i'

#define BATCH   16384
#define IN_DIM  1024
#define X1D     512
#define HID     2048

#define TM 128
#define TN 128
#define TK 64
#define LDSK 72   // padded LDS row stride in bf16 elems (144 B: 16B-aligned, bank-conflict-free)

#define GLOBAL_AS __attribute__((address_space(1)))
#define LDS_AS    __attribute__((address_space(3)))

#if defined(__gfx1250__) && __has_builtin(__builtin_amdgcn_global_load_async_to_lds_b128)
#define USE_ASYNC_LDS 1
#else
#define USE_ASYNC_LDS 0
#endif

#if USE_ASYNC_LDS
#if __has_builtin(__builtin_amdgcn_s_wait_asynccnt)
#define WAIT_ASYNC(n) __builtin_amdgcn_s_wait_asynccnt(n)
#else
#define WAIT_ASYNC(n) asm volatile("s_wait_asynccnt %0" ::"i"(n) : "memory")
#endif
#else
#define WAIT_ASYNC(n)
#endif

// Copy 16 bytes global -> LDS. Async path uses the gfx1250 ASYNC-LDS pipe
// (global_load_async_to_lds_b128, tracked by ASYNCcnt).
__device__ __forceinline__ void cp16_g2l(const __bf16* g, __bf16* l) {
#if USE_ASYNC_LDS
    __builtin_amdgcn_global_load_async_to_lds_b128(
        (GLOBAL_AS v4i_gcc*)(g), (LDS_AS v4i_gcc*)(l), 0, 0);
#else
    *reinterpret_cast<uint4*>(l) = *reinterpret_cast<const uint4*>(g);
#endif
}

__device__ __forceinline__ __bf16 f2bf(float f) {
    unsigned int u = __builtin_bit_cast(unsigned int, f);
    u = (u + 0x7FFFu + ((u >> 16) & 1u)) >> 16;   // round-to-nearest-even
    unsigned short h = (unsigned short)u;
    return __builtin_bit_cast(__bf16, h);
}

// Load one 16x32 bf16 WMMA operand fragment from an LDS tile (row-major, K-major).
// Per ISA 7.12.2: lanes 0-15 hold K[0..7] (elems 0-7) and K[16..23] (elems 8-15),
// lanes 16-31 hold K[8..15] and K[24..31]. col = ks*32 + half*8.
__device__ __forceinline__ v16bf ldfrag(const __bf16* base, int row, int col) {
    union { v16bf v; uint4 u[2]; } f;
    const __bf16* p = base + row * LDSK + col;
    f.u[0] = *reinterpret_cast<const uint4*>(p);        // K-block lo (16 B)
    f.u[1] = *reinterpret_cast<const uint4*>(p + 16);   // K-block hi (+32 B)
    return f.v;
}

// ---------------------------------------------------------------------------
// Conversion kernels
// ---------------------------------------------------------------------------

// x[:, :512] (f32, row stride 1024) -> bf16 (row stride 512)
__global__ void convert_x1_kernel(const float* __restrict__ x, __bf16* __restrict__ xbf) {
    int idx = blockIdx.x * 256 + threadIdx.x;     // BATCH*512 total
    int row = idx >> 9, col = idx & 511;
    xbf[idx] = f2bf(x[(size_t)row * IN_DIM + col]);
}

// w (K x N, f32, row-major) -> wT (N x K, bf16, row-major)  [B fed column-major to WMMA]
__global__ void convert_wT_kernel(const float* __restrict__ w, __bf16* __restrict__ wT,
                                  int K, int N) {
    __shared__ float tile[32][33];
    int bk = blockIdx.x * 32, bn = blockIdx.y * 32;
    int tx = threadIdx.x & 31, ty = threadIdx.x >> 5;   // 256 thr: ty 0..7
#pragma unroll
    for (int i = 0; i < 32; i += 8)
        tile[ty + i][tx] = w[(size_t)(bk + ty + i) * N + bn + tx];
    __syncthreads();
#pragma unroll
    for (int i = 0; i < 32; i += 8)
        wT[(size_t)(bn + ty + i) * K + bk + tx] = f2bf(tile[tx][ty + i]);
}

// ---------------------------------------------------------------------------
// Fused GEMM:  Out = epilogue(A[M,K] @ BT[N,K]^T + bias)
//   MODE 0: relu((acc+bias)*exp(g)+c) -> bf16
//   MODE 1: tanh(acc+bias)            -> f32   (scale head)
//   MODE 2: acc+bias                  -> f32   (translate head)
// 256 threads = 8 waves (4M x 2N); each wave: 32x64 = 2x4 WMMA tiles.
// ---------------------------------------------------------------------------
template <int MODE>
__global__ void __launch_bounds__(256) gemm_bf16_kernel(
    const __bf16* __restrict__ A,   // M x K
    const __bf16* __restrict__ BT,  // N x K
    const float*  __restrict__ bias,
    const float*  __restrict__ gn,
    const float*  __restrict__ cn,
    void*         __restrict__ OutP,
    int N, int K)
{
    __shared__ __bf16 lA[2][TM * LDSK];
    __shared__ __bf16 lB[2][TN * LDSK];

    const int tid  = threadIdx.x;
    const int lane = tid & 31;
    const int wave = tid >> 5;
    const int wm   = wave & 3;        // 0..3  -> row offset wm*32
    const int wn   = wave >> 2;       // 0..1  -> col offset wn*64
    const int lr   = lane & 15;
    const int half = lane >> 4;

    const int mBase = blockIdx.y * TM;
    const int nBase = blockIdx.x * TN;

    // global->LDS tile map: 1024 16B chunks (128 rows x 8), 4 per thread
    const int arow = tid >> 3;        // 0..31, +j*32
    const int acol = tid & 7;         // 16B chunk within 128B row

    v8f acc[2][4];
#pragma unroll
    for (int mi = 0; mi < 2; ++mi)
#pragma unroll
        for (int ni = 0; ni < 4; ++ni)
#pragma unroll
            for (int e = 0; e < 8; ++e) acc[mi][ni][e] = 0.0f;

    // Issue the 8 async 16B copies that bring one K-tile (A+B halves) in.
    auto issueTile = [&](int buf, int kBase) {
#pragma unroll
        for (int j = 0; j < 4; ++j) {
            const int r = arow + j * 32;
            cp16_g2l(A  + (size_t)(mBase + r) * K + kBase + acol * 8,
                     &lA[buf][r * LDSK + acol * 8]);
            cp16_g2l(BT + (size_t)(nBase + r) * K + kBase + acol * 8,
                     &lB[buf][r * LDSK + acol * 8]);
        }
    };

    auto computeTile = [&](int buf) {
#pragma unroll
        for (int ks = 0; ks < 2; ++ks) {           // two K=32 steps per K-tile
            v16bf af[2], bfm[4];
#pragma unroll
            for (int mi = 0; mi < 2; ++mi)
                af[mi] = ldfrag(&lA[buf][0], wm * 32 + mi * 16 + lr, ks * 32 + half * 8);
#pragma unroll
            for (int ni = 0; ni < 4; ++ni)
                bfm[ni] = ldfrag(&lB[buf][0], wn * 64 + ni * 16 + lr, ks * 32 + half * 8);
#pragma unroll
            for (int mi = 0; mi < 2; ++mi)
#pragma unroll
                for (int ni = 0; ni < 4; ++ni)
                    acc[mi][ni] = __builtin_amdgcn_wmma_f32_16x16x32_bf16(
                        false, af[mi], false, bfm[ni],
                        (short)0, acc[mi][ni], false, false);
        }
    };

    const int nk = K / TK;
    issueTile(0, 0);
    for (int kt = 0; kt < nk; ++kt) {
        const int cur = kt & 1;
        if (kt + 1 < nk) {
            issueTile(cur ^ 1, (kt + 1) * TK);   // prefetch next tile (other buffer)
            WAIT_ASYNC(8);                       // in-order: current tile's 8 done
        } else {
            WAIT_ASYNC(0);
        }
        __syncthreads();                         // all waves' copies visible
        computeTile(cur);
        __syncthreads();                         // done reading before next overwrite
    }

    // Epilogue. C layout (7.12.2): VGPR e, lanes 0-15: M=e, N=lane; lanes 16-31: M=e+8.
#pragma unroll
    for (int ni = 0; ni < 4; ++ni) {
        const int gc = nBase + wn * 64 + ni * 16 + lr;   // lane-invariant over e
        const float bb = bias[gc];
        float eg = 0.f, cc = 0.f;
        if (MODE == 0) { eg = __expf(gn[gc]); cc = cn[gc]; }
#pragma unroll
        for (int mi = 0; mi < 2; ++mi) {
#pragma unroll
            for (int e = 0; e < 8; ++e) {
                const int gr = mBase + wm * 32 + mi * 16 + half * 8 + e;
                float v = acc[mi][ni][e] + bb;
                if (MODE == 0) {
                    v = fmaxf(v * eg + cc, 0.0f);
                    reinterpret_cast<__bf16*>(OutP)[(size_t)gr * N + gc] = f2bf(v);
                } else if (MODE == 1) {
                    reinterpret_cast<float*>(OutP)[(size_t)gr * N + gc] = tanhf(v);
                } else {
                    reinterpret_cast<float*>(OutP)[(size_t)gr * N + gc] = v;
                }
            }
        }
    }
}

// ---------------------------------------------------------------------------
// Final fuse: h2*exp(scale)+translate, output actnorm, per-row log-det.
// One block per row.
// ---------------------------------------------------------------------------
__global__ void __launch_bounds__(256) combine_kernel(
    const float* __restrict__ x, const float* __restrict__ scale,
    const float* __restrict__ trans, const float* __restrict__ og,
    const float* __restrict__ oc, float* __restrict__ out,
    float* __restrict__ logdet)
{
    const int b = blockIdx.x;
    const int t = threadIdx.x;
    float local = 0.0f;
#pragma unroll
    for (int j = t; j < IN_DIM; j += 256) local += og[j];     // sum(og)
    for (int j = t; j < X1D; j += 256) {
        const float s  = scale[(size_t)b * X1D + j];
        local += s;
        const float tr = trans[(size_t)b * X1D + j];
        const float h2 = x[(size_t)b * IN_DIM + X1D + j];
        const float v  = h2 * __expf(s) + tr;
        out[(size_t)b * IN_DIM + X1D + j] = v * __expf(og[X1D + j]) + oc[X1D + j];
        const float h1 = x[(size_t)b * IN_DIM + j];
        out[(size_t)b * IN_DIM + j] = h1 * __expf(og[j]) + oc[j];
    }
    __shared__ float red[256];
    red[t] = local;
    __syncthreads();
#pragma unroll
    for (int s = 128; s > 0; s >>= 1) {
        if (t < s) red[t] += red[t + s];
        __syncthreads();
    }
    if (t == 0) logdet[b] = red[0];
}

// ---------------------------------------------------------------------------
// Host side
// ---------------------------------------------------------------------------
extern "C" void kernel_launch(void* const* d_in, const int* in_sizes, int n_in,
                              void* d_out, int out_size, void* d_ws, size_t ws_size,
                              hipStream_t stream) {
    (void)in_sizes; (void)n_in; (void)out_size; (void)ws_size;

    const float* x  = (const float*)d_in[0];
    const float* og = (const float*)d_in[1];
    const float* oc = (const float*)d_in[2];
    const float* S[14]; for (int i = 0; i < 14; ++i) S[i] = (const float*)d_in[3 + i];
    const float* T[14]; for (int i = 0; i < 14; ++i) T[i] = (const float*)d_in[17 + i];

    char* ws = (char*)d_ws;
    __bf16* Xbf  = (__bf16*)ws; ws += (size_t)BATCH * X1D * 2;      // 16 MB
    __bf16* w1T  = (__bf16*)ws; ws += (size_t)HID * X1D * 2;        //  2 MB
    __bf16* w2T  = (__bf16*)ws; ws += (size_t)HID * HID * 2;        //  8 MB
    __bf16* w3T  = (__bf16*)ws; ws += (size_t)HID * HID * 2;        //  8 MB
    __bf16* w4T  = (__bf16*)ws; ws += (size_t)X1D * HID * 2;        //  2 MB
    __bf16* actA = (__bf16*)ws; ws += (size_t)BATCH * HID * 2;      // 64 MB
    __bf16* actB = (__bf16*)ws; ws += (size_t)BATCH * HID * 2;      // 64 MB
    float* scaleF = (float*)ws; ws += (size_t)BATCH * X1D * 4;      // 32 MB
    float* transF = (float*)ws;                                     // 32 MB

    float* out    = (float*)d_out;
    float* logdet = out + (size_t)BATCH * IN_DIM;

    convert_x1_kernel<<<(BATCH * X1D) / 256, 256, 0, stream>>>(x, Xbf);

    auto runMLP = [&](const float* const* W, float* finalOut, int finalMode) {
        convert_wT_kernel<<<dim3(X1D / 32, HID / 32), 256, 0, stream>>>(W[0],  w1T, X1D, HID);
        convert_wT_kernel<<<dim3(HID / 32, HID / 32), 256, 0, stream>>>(W[4],  w2T, HID, HID);
        convert_wT_kernel<<<dim3(HID / 32, HID / 32), 256, 0, stream>>>(W[8],  w3T, HID, HID);
        convert_wT_kernel<<<dim3(HID / 32, X1D / 32), 256, 0, stream>>>(W[12], w4T, HID, X1D);

        gemm_bf16_kernel<0><<<dim3(HID / TN, BATCH / TM), 256, 0, stream>>>(
            Xbf,  w1T, W[1],  W[2],  W[3],  actA, HID, X1D);
        gemm_bf16_kernel<0><<<dim3(HID / TN, BATCH / TM), 256, 0, stream>>>(
            actA, w2T, W[5],  W[6],  W[7],  actB, HID, HID);
        gemm_bf16_kernel<0><<<dim3(HID / TN, BATCH / TM), 256, 0, stream>>>(
            actB, w3T, W[9],  W[10], W[11], actA, HID, HID);
        if (finalMode == 1)
            gemm_bf16_kernel<1><<<dim3(X1D / TN, BATCH / TM), 256, 0, stream>>>(
                actA, w4T, W[13], nullptr, nullptr, finalOut, X1D, HID);
        else
            gemm_bf16_kernel<2><<<dim3(X1D / TN, BATCH / TM), 256, 0, stream>>>(
                actA, w4T, W[13], nullptr, nullptr, finalOut, X1D, HID);
    };

    runMLP(S, scaleF, 1);   // scale net (tanh head)
    runMLP(T, transF, 2);   // translate net

    combine_kernel<<<BATCH, 256, 0, stream>>>(x, scaleF, transF, og, oc, out, logdet);
}
